// Head_12506944766097
// MI455X (gfx1250) — matrix operand order repaired and checked by reading, not compiled, not run on hardware
//
#include <hip/hip_runtime.h>

// ---------------------------------------------------------------------------
// Single-head causal attention for MI455X (gfx1250, wave32, WMMA).
// B=8, T=2048, D=1024, H=64.  bf16 matrix math, fp32 accumulate + softmax.
// ---------------------------------------------------------------------------

typedef __attribute__((ext_vector_type(16))) __bf16 v16bf;
typedef __attribute__((ext_vector_type(8)))  float  v8f;

#define B_  8
#define T_  2048
#define D_  1024
#define H_  64
#define BTH (B_ * T_ * H_)   // 1048576 elements per q/k/v tensor

union BF16Frag {
    v16bf          v;
    unsigned int   u32[8];
    uint4          u128[2];
    unsigned short u16[16];
};

__device__ __forceinline__ unsigned short f2bf(float f) {
    unsigned int u = __float_as_uint(f);
    unsigned int r = u + 0x7FFFu + ((u >> 16) & 1u);   // round-to-nearest-even
    return (unsigned short)(r >> 16);
}

__device__ __forceinline__ v8f vzero8() {
    v8f z;
#pragma unroll
    for (int i = 0; i < 8; ++i) z[i] = 0.0f;
    return z;
}

__device__ __forceinline__ v8f wmma_bf16(const BF16Frag& a, const BF16Frag& b, v8f c) {
    return __builtin_amdgcn_wmma_f32_16x16x32_bf16(
        /*neg_a=*/false, a.v, /*neg_b=*/false, b.v,
        /*c_mod=*/(short)0, c, /*reuse_a=*/false, /*reuse_b=*/false);
}

// ---------------------------------------------------------------------------
// Kernel 0: convert Wq/Wk/Wv fp32 -> bf16, permuted into WMMA B-fragment
// order.  Fragment element i of lane L for (kc, nt) lives at
//   wperm[s*65536 + ((kc*4 + nt)*32 + L)*16 + i]
// and equals W[K*H + N] with K = kc*32 + 16*(L/16) + i, N = nt*16 + (L%16).
// ---------------------------------------------------------------------------
__global__ __launch_bounds__(256) void wperm_kernel(const float* __restrict__ Wq,
                                                    const float* __restrict__ Wk,
                                                    const float* __restrict__ Wv,
                                                    unsigned short* __restrict__ wperm) {
    int t    = blockIdx.x * 256 + threadIdx.x;   // 0 .. 3*65536-1
    int wsel = t >> 16;
    int o    = t & 65535;
    int i    = o & 15;
    int L    = (o >> 4) & 31;
    int nt   = (o >> 9) & 3;
    int kc   = o >> 11;
    int K    = kc * 32 + ((L >> 4) << 4) + i;
    int N    = nt * 16 + (L & 15);
    const float* W = (wsel == 0) ? Wq : (wsel == 1) ? Wk : Wv;
    wperm[t] = f2bf(W[K * H_ + N]);
}

// ---------------------------------------------------------------------------
// Kernel 1: fused QKV projection.  One block = 128 threads (4 waves) = 64
// rows of x.  Each wave owns 16 rows and all 64 output columns of q, k, v.
// All 12 B-fragments are preloaded per K-chunk so the compiler can clause
// the global_load_b128s and overlap them with the WMMA stream.
// ---------------------------------------------------------------------------
__global__ __launch_bounds__(128) void qkv_kernel(const float* __restrict__ x,
                                                  const unsigned short* __restrict__ wperm,
                                                  unsigned short* __restrict__ qkv) {
    __shared__ __align__(16) unsigned short xbuf[64 * 32];   // 64 rows x 32 K, bf16

    const int tid  = threadIdx.x;
    const int wave = tid >> 5;
    const int lane = tid & 31;
    const int half = lane >> 4;
    const int lan  = lane & 15;
    const long row_base = (long)blockIdx.x * 64;

    v8f acc[12];
#pragma unroll
    for (int i = 0; i < 12; ++i) acc[i] = vzero8();

    for (int kc = 0; kc < 32; ++kc) {
        // preload all 12 W B-fragments (independent of the LDS barrier)
        BF16Frag wb[12];
#pragma unroll
        for (int f = 0; f < 12; ++f) {
            int s  = f >> 2;
            int nt = f & 3;
            const unsigned short* p =
                wperm + s * 65536 + (((kc * 4 + nt) * 32 + lane) << 4);
            wb[f].u128[0] = *(const uint4*)p;
            wb[f].u128[1] = *(const uint4*)(p + 8);
        }

        // stage 64x32 fp32 -> bf16 into LDS (float4 loads, uint2 stores)
#pragma unroll
        for (int i = 0; i < 4; ++i) {
            int f4  = tid + i * 128;       // 0..511 float4s
            int row = f4 >> 3;
            int c4  = (f4 & 7) * 4;
            float4 vl = *(const float4*)(x + (row_base + row) * D_ + kc * 32 + c4);
            unsigned int lo = (unsigned int)f2bf(vl.x) | ((unsigned int)f2bf(vl.y) << 16);
            unsigned int hi = (unsigned int)f2bf(vl.z) | ((unsigned int)f2bf(vl.w) << 16);
            *(uint2*)&xbuf[row * 32 + c4] = make_uint2(lo, hi);
        }
        __syncthreads();

        // A fragment (16x32 bf16): lane holds row (wave*16 + lan),
        // K runs [16*half .. +15]
        BF16Frag xa;
        {
            const unsigned short* p = &xbuf[(wave * 16 + lan) * 32 + 16 * half];
            xa.u128[0] = *(const uint4*)p;
            xa.u128[1] = *(const uint4*)(p + 8);
        }

#pragma unroll
        for (int f = 0; f < 12; ++f)
            acc[f] = wmma_bf16(xa, wb[f], acc[f]);
        __syncthreads();
    }

    // store q/k/v bf16 row-major [B*T][64]
#pragma unroll
    for (int s = 0; s < 3; ++s) {
        unsigned short* dst = qkv + (size_t)s * BTH;
#pragma unroll
        for (int r = 0; r < 8; ++r) {
            long row = row_base + wave * 16 + r + 8 * half;
#pragma unroll
            for (int nt = 0; nt < 4; ++nt)
                dst[row * H_ + nt * 16 + lan] = f2bf(acc[s * 4 + nt][r]);
        }
    }
}

// ---------------------------------------------------------------------------
// Kernel 2: flash attention.  One block = 128 threads (4 waves) = 64 query
// rows of one batch.  K blocks are copied memory->LDS with the CDNA5 async
// path (GLOBAL_LOAD_ASYNC_TO_LDS_B128, ASYNCcnt); V is transposed through
// VGPRs so P@V B-fragments are contiguous LDS reads.
// ---------------------------------------------------------------------------
__global__ __launch_bounds__(128) void attn_kernel(const unsigned short* __restrict__ qkv,
                                                   float* __restrict__ out) {
    __shared__ __align__(16) unsigned short kbuf[32 * 64];      // [key][h] bf16
    __shared__ __align__(16) unsigned short vtb[64 * 32];       // [h][key] bf16
    __shared__ __align__(16) unsigned short pbuf[4 * 16 * 32];  // per-wave P scratch

    const int tid  = threadIdx.x;
    const int wave = tid >> 5;
    const int lane = tid & 31;
    const int half = lane >> 4;
    const int lan  = lane & 15;

    const int  tilesPerB = T_ / 64;
    const int  b      = blockIdx.x / tilesPerB;
    const int  q_base = (blockIdx.x % tilesPerB) * 64;
    const long rowQ   = (long)b * T_ + q_base;

    const unsigned short* qp = qkv;
    const unsigned short* kp = qkv + (size_t)BTH;
    const unsigned short* vp = qkv + 2 * (size_t)BTH;

    // fold 1/sqrt(H) and log2(e) into one scale; softmax done in base 2
    const float scale2 = 0.125f * 1.44269504088896f;

    // persistent q A-fragments (H=64 -> two K=32 chunks)
    BF16Frag qa[2];
    {
        const unsigned short* qrow = qp + (rowQ + wave * 16 + lan) * H_;
#pragma unroll
        for (int c = 0; c < 2; ++c) {
            qa[c].u128[0] = *(const uint4*)(qrow + c * 32 + 16 * half);
            qa[c].u128[1] = *(const uint4*)(qrow + c * 32 + 16 * half + 8);
        }
    }

    float mst[8], lst[8];
    v8f   acc[4];
#pragma unroll
    for (int r = 0; r < 8; ++r) { mst[r] = -1.0e30f; lst[r] = 0.0f; }
#pragma unroll
    for (int nt = 0; nt < 4; ++nt) acc[nt] = vzero8();

    const unsigned int lds_k = (unsigned int)(size_t)(&kbuf[0]);

    const int nkb = (q_base + 64 + 31) / 32;   // key blocks needed (uniform)
    for (int kb = 0; kb < nkb; ++kb) {
        const long rowK = (long)b * T_ + kb * 32;

        // ---- async-copy K block straight into LDS (no VGPR round trip) ----
        {
            const unsigned long long kgbase =
                (unsigned long long)(size_t)(kp + rowK * H_);
#pragma unroll
            for (int i = 0; i < 2; ++i) {
                unsigned int bo = (unsigned int)(tid + i * 128) * 16u;  // byte off
                asm volatile("global_load_async_to_lds_b128 %0, %1, %2"
                             :: "v"(lds_k + bo), "v"(bo), "s"(kgbase)
                             : "memory");
            }
        }
        // ---- V block: load row-major uint4, write transposed into vtb ----
        {
            const uint4* vs = (const uint4*)(vp + rowK * H_);
#pragma unroll
            for (int i = 0; i < 2; ++i) {
                int u4  = tid + i * 128;       // 0..255 uint4s
                int key = u4 >> 3;
                int h8  = (u4 & 7) * 8;
                uint4 vv = vs[u4];
                unsigned int w[4] = {vv.x, vv.y, vv.z, vv.w};
#pragma unroll
                for (int j = 0; j < 4; ++j) {
                    vtb[(h8 + 2 * j) * 32 + key]     = (unsigned short)(w[j] & 0xFFFFu);
                    vtb[(h8 + 2 * j + 1) * 32 + key] = (unsigned short)(w[j] >> 16);
                }
            }
        }
        if (kb + 1 < nkb)
            __builtin_prefetch(kp + (rowK + 32) * H_ + tid * 8, 0, 0);
        asm volatile("s_wait_asynccnt 0x0" ::: "memory");
        __syncthreads();

        // ---- S = q @ k^T  (two 16x16 tiles over 32 keys) ----
        v8f s0 = vzero8(), s1 = vzero8();
#pragma unroll
        for (int c = 0; c < 2; ++c) {
            BF16Frag bk0, bk1;
            const unsigned short* p0 = &kbuf[lan * 64 + c * 32 + 16 * half];
            bk0.u128[0] = *(const uint4*)p0;
            bk0.u128[1] = *(const uint4*)(p0 + 8);
            const unsigned short* p1 = &kbuf[(16 + lan) * 64 + c * 32 + 16 * half];
            bk1.u128[0] = *(const uint4*)p1;
            bk1.u128[1] = *(const uint4*)(p1 + 8);
            s0 = wmma_bf16(qa[c], bk0, s0);
            s1 = wmma_bf16(qa[c], bk1, s1);
        }

        // ---- causal mask + online softmax (rows live per lane-half) ----
        const int qg  = q_base + wave * 16 + 8 * half;   // row = qg + r
        const int kg0 = kb * 32 + lan;
        const int kg1 = kg0 + 16;
#pragma unroll
        for (int r = 0; r < 8; ++r) {
            int   qrow = qg + r;
            float v0 = (kg0 <= qrow) ? s0[r] * scale2 : -1.0e30f;
            float v1 = (kg1 <= qrow) ? s1[r] * scale2 : -1.0e30f;
            float mx = fmaxf(v0, v1);
#pragma unroll
            for (int m = 8; m >= 1; m >>= 1)
                mx = fmaxf(mx, __shfl_xor(mx, m, 32));
            float mn   = fmaxf(mst[r], mx);
            float corr = exp2f(mst[r] - mn);
            float p0f  = exp2f(v0 - mn);
            float p1f  = exp2f(v1 - mn);
            float sum  = p0f + p1f;
#pragma unroll
            for (int m = 8; m >= 1; m >>= 1)
                sum += __shfl_xor(sum, m, 32);
            lst[r] = lst[r] * corr + sum;
            mst[r] = mn;
#pragma unroll
            for (int nt = 0; nt < 4; ++nt) acc[nt][r] = acc[nt][r] * corr;

            int mrow = r + 8 * half;
            pbuf[wave * 512 + mrow * 32 + lan]      = f2bf(p0f);
            pbuf[wave * 512 + mrow * 32 + 16 + lan] = f2bf(p1f);
        }
        __syncthreads();

        // ---- rebuild P as A-fragment (16x32) ----
        BF16Frag pa;
#pragma unroll
        for (int jj = 0; jj < 8; ++jj) {
            int K = 16 * half + 2 * jj;
            pa.u32[jj] = *(const unsigned int*)&pbuf[wave * 512 + lan * 32 + K];
        }

        // ---- O += P @ V ----
#pragma unroll
        for (int nt = 0; nt < 4; ++nt) {
            BF16Frag bv;
            const unsigned short* p = &vtb[(nt * 16 + lan) * 32 + 16 * half];
            bv.u128[0] = *(const uint4*)p;
            bv.u128[1] = *(const uint4*)(p + 8);
            acc[nt] = wmma_bf16(pa, bv, acc[nt]);
        }
        __syncthreads();   // protect kbuf/vtb before next staging
    }

    // ---- normalize and store fp32 output ----
#pragma unroll
    for (int r = 0; r < 8; ++r) {
        long  row = rowQ + wave * 16 + r + 8 * half;
        float rl  = 1.0f / lst[r];
#pragma unroll
        for (int nt = 0; nt < 4; ++nt)
            out[row * H_ + nt * 16 + lan] = acc[nt][r] * rl;
    }
}

// ---------------------------------------------------------------------------
extern "C" void kernel_launch(void* const* d_in, const int* in_sizes, int n_in,
                              void* d_out, int out_size, void* d_ws, size_t ws_size,
                              hipStream_t stream) {
    (void)in_sizes; (void)n_in; (void)out_size; (void)ws_size;
    const float* x  = (const float*)d_in[0];
    const float* Wq = (const float*)d_in[1];
    const float* Wk = (const float*)d_in[2];
    const float* Wv = (const float*)d_in[3];

    unsigned short* ws    = (unsigned short*)d_ws;
    unsigned short* wperm = ws;               // 3 * 65536 bf16
    unsigned short* qkv   = ws + 3 * 65536;   // 3 * B*T*H bf16

    wperm_kernel<<<768, 256, 0, stream>>>(Wq, Wk, Wv, wperm);
    qkv_kernel<<<(B_ * T_) / 64, 128, 0, stream>>>(x, wperm, qkv);
    attn_kernel<<<B_ * (T_ / 64), 128, 0, stream>>>(qkv, (float*)d_out);
}
